// Edge_embedding_tetris_88656714925207
// MI455X (gfx1250) — matrix operand order repaired and checked by reading, not compiled, not run on hardware
//
#include <hip/hip_runtime.h>
#include <math.h>

typedef __attribute__((ext_vector_type(16))) _Float16 v16h;
typedef __attribute__((ext_vector_type(8)))  float    v8f;
typedef __attribute__((ext_vector_type(2)))  _Float16 h2t;

#define AADD(p, x) __hip_atomic_fetch_add((p), (x), __ATOMIC_RELAXED, __HIP_MEMORY_SCOPE_AGENT)

__global__ void zero_f32(float* __restrict__ p, int n) {
  int i = blockIdx.x * blockDim.x + threadIdx.x;
  if (i < n) p[i] = 0.0f;
}

// Build the 16x32 f16 A-matrix fragment (one K-chunk) for this lane from the
// pre-swizzled (w1,b1) half2 table: h = relu(vn * w1[K] + b1[K]).
__device__ __forceinline__ v16h build_a(const uint2* wb, h2t vn2) {
  v16h a;
#pragma unroll
  for (int p = 0; p < 8; ++p) {
    uint2 q = wb[p];
    h2t w = __builtin_bit_cast(h2t, q.x);
    h2t b = __builtin_bit_cast(h2t, q.y);
    h2t h = w * vn2 + b;                 // v_pk_fma_f16
    _Float16 z = (_Float16)0.0f;
    a[2 * p]     = (h[0] > z) ? h[0] : z;
    a[2 * p + 1] = (h[1] > z) ? h[1] : z;
  }
  return a;
}

__launch_bounds__(256, 4)
__global__ void edge_mlp_scatter(
    const float* __restrict__ v, const float* __restrict__ rot,
    const int* __restrict__ edge_index,
    const float* __restrict__ msg_w1, const float* __restrict__ msg_b1,
    const float* __restrict__ msg_w2, const float* __restrict__ msg_b2,
    const float* __restrict__ sc_w1,  const float* __restrict__ sc_b1,
    const float* __restrict__ sc_w2,  const float* __restrict__ sc_b2,
    float* __restrict__ out, int E, int N) {
  // (w1,b1) pairs pre-swizzled into per-lane WMMA A-layout order: [mlp][chunk][lane][pair]
  __shared__ uint2 sWB[2][3][32][8];                       // 12 KB
  // w2 pre-swizzled into per-lane WMMA B-layout order: [mlp][chunk][ntile][lane][i]
  __shared__ __align__(32) _Float16 sB[2][3][2][32][16];   // 12 KB

  for (int idx = threadIdx.x; idx < 1536; idx += 256) {
    int p = idx & 7, L = (idx >> 3) & 31;
    int c = (idx >> 8) % 3, m = idx / 768;
    const float* w1 = m ? sc_w1 : msg_w1;
    const float* b1 = m ? sc_b1 : msg_b1;
    // A-layout K index: K = 32c + 8*(L/16) + 2p (+8 if p>=4)
    int K = 32 * c + ((L >> 4) << 3) + 2 * p + ((p >= 4) ? 8 : 0);
    h2t wp = { (_Float16)w1[K], (_Float16)w1[K + 1] };
    h2t bp = { (_Float16)b1[K], (_Float16)b1[K + 1] };
    uint2 u;
    u.x = __builtin_bit_cast(unsigned, wp);
    u.y = __builtin_bit_cast(unsigned, bp);
    sWB[m][c][L][p] = u;
  }
  for (int idx = threadIdx.x; idx < 6144; idx += 256) {
    int i = idx & 15, L = (idx >> 4) & 31, t = (idx >> 9) & 1;
    int c = (idx >> 10) % 3, m = idx / 3072;
    const float* w2 = m ? sc_w2 : msg_w2;
    // B-layout: lane L holds N = L%16, K = 32c + 8*(L/16) + i (+8 if i>=8)
    int K  = 32 * c + ((L >> 4) << 3) + i + ((i >= 8) ? 8 : 0);
    int Nc = 16 * t + (L & 15);
    sB[m][c][t][L][i] = (_Float16)w2[K * 32 + Nc];
  }
  __syncthreads();

  const int lane = threadIdx.x & 31;
  const int g    = lane >> 4;
  const int m15  = lane & 15;
  const int nTiles = (E + 15) >> 4;
  const int waveId = blockIdx.x * 8 + (threadIdx.x >> 5);
  const int nWaves = gridDim.x * 8;
  const int* col = edge_index + E;      // row 1 of [2,E]
  float* outS = out;                    // scalar_features [N,32]
  float* outR = out + (size_t)N * 32;   // rot_features    [N,16,2] == [N,32]

  const float bM0 = msg_b2[m15], bM1 = msg_b2[16 + m15];
  const float bS0 = sc_b2[m15],  bS1 = sc_b2[16 + m15];

  for (int tile = waveId; tile < nTiles; tile += nWaves) {
    // Launder the LDS base offset so the weight-table loads are NOT
    // loop-invariant to the optimizer: keeps them as in-loop ds_load_b64/b128
    // instead of being hoisted into ~200 live VGPRs (which blew past 256 regs
    // and caused s_set_vgpr_msb thrash + 1-wave occupancy).
    unsigned zOff = 0;
    asm volatile("" : "+v"(zOff));
    const uint2*    wbT = (const uint2*)((const char*)&sWB[0][0][0][0] + zOff);
    const _Float16* bT  = (const _Float16*)((const char*)&sB[0][0][0][0][0] + zOff);

    const int base = tile << 4;
    int eA = base + m15; if (eA >= E) eA = E - 1;     // clamp (tail safety)
    float vx = v[2 * eA], vy = v[2 * eA + 1];
    float vn = sqrtf(vx * vx + vy * vy);
    float cth = rot[4 * eA];         // cos
    float sth = rot[4 * eA + 2];     // sin
    int   cl  = col[eA];
    _Float16 hn = (_Float16)vn;
    h2t vn2 = { hn, hn };

    v8f aM0 = { bM0, bM0, bM0, bM0, bM0, bM0, bM0, bM0 };
    v8f aM1 = { bM1, bM1, bM1, bM1, bM1, bM1, bM1, bM1 };
    v8f aS0 = { bS0, bS0, bS0, bS0, bS0, bS0, bS0, bS0 };
    v8f aS1 = { bS1, bS1, bS1, bS1, bS1, bS1, bS1, bS1 };

#pragma unroll
    for (int c = 0; c < 3; ++c) {
      const uint2* wbM = wbT + ((0 * 3 + c) * 32 + lane) * 8;
      const uint2* wbS = wbT + ((1 * 3 + c) * 32 + lane) * 8;
      v16h A  = build_a(wbM, vn2);
      v16h B0 = *(const v16h*)(bT + (((0 * 3 + c) * 2 + 0) * 32 + lane) * 16);
      v16h B1 = *(const v16h*)(bT + (((0 * 3 + c) * 2 + 1) * 32 + lane) * 16);
      aM0 = __builtin_amdgcn_wmma_f32_16x16x32_f16(false, A, false, B0, (short)0, aM0, false, false);
      aM1 = __builtin_amdgcn_wmma_f32_16x16x32_f16(false, A, false, B1, (short)0, aM1, false, false);
      v16h As = build_a(wbS, vn2);
      v16h B2 = *(const v16h*)(bT + (((1 * 3 + c) * 2 + 0) * 32 + lane) * 16);
      v16h B3 = *(const v16h*)(bT + (((1 * 3 + c) * 2 + 1) * 32 + lane) * 16);
      aS0 = __builtin_amdgcn_wmma_f32_16x16x32_f16(false, As, false, B2, (short)0, aS0, false, false);
      aS1 = __builtin_amdgcn_wmma_f32_16x16x32_f16(false, As, false, B3, (short)0, aS1, false, false);
    }

    // D layout: VGPR r -> edge row (r + 8g), channel N = m15 (acc0) / 16+m15 (acc1).
#pragma unroll
    for (int r = 0; r < 8; ++r) {
      int src = r + 8 * g;                    // edge row within tile
      int   ci = __shfl(cl,  src, 32);
      float cc = __shfl(cth, src, 32);
      float ss = __shfl(sth, src, 32);
      float mA = aM0[r], mB = aM1[r];
      float pA = __shfl_xor(mA, 1, 32);       // partner channel (2j <-> 2j+1)
      float pB = __shfl_xor(mB, 1, 32);
      float sg = (lane & 1) ? -1.0f : 1.0f;   // l=0: +m1*s ; l=1: -m0*s
      float rA = mA * cc + sg * (pA * ss);
      float rB = mB * cc + sg * (pB * ss);
      if (base + src < E) {
        size_t o = (size_t)ci * 32;
        AADD(&outS[o + m15],      aS0[r]);
        AADD(&outS[o + 16 + m15], aS1[r]);
        AADD(&outR[o + m15],      rA);
        AADD(&outR[o + 16 + m15], rB);
      }
    }
  }
}

extern "C" void kernel_launch(void* const* d_in, const int* in_sizes, int n_in,
                              void* d_out, int out_size, void* d_ws, size_t ws_size,
                              hipStream_t stream) {
  const float* v    = (const float*)d_in[0];
  const float* rot  = (const float*)d_in[1];
  const int*   ei   = (const int*)d_in[2];
  const float* mw1  = (const float*)d_in[3];
  const float* mb1  = (const float*)d_in[4];
  const float* mw2  = (const float*)d_in[5];
  const float* mb2  = (const float*)d_in[6];
  const float* sw1  = (const float*)d_in[7];
  const float* sb1  = (const float*)d_in[8];
  const float* sw2  = (const float*)d_in[9];
  const float* sb2  = (const float*)d_in[10];
  float* out = (float*)d_out;

  const int E = in_sizes[0] / 2;     // v is [E,2]
  const int N = out_size / 64;       // outputs are [N,32] + [N,32]

  int zb = (out_size + 255) / 256;
  zero_f32<<<zb, 256, 0, stream>>>(out, out_size);

  // 1250 blocks x 8 waves = 10000 waves; 100k tiles -> ~10 tiles/wave,
  // amortizing the per-block LDS weight swizzle over ~80 tiles.
  edge_mlp_scatter<<<1250, 256, 0, stream>>>(v, rot, ei, mw1, mb1, mw2, mb2,
                                             sw1, sb1, sw2, sb2, out, E, N);
}
